// MLP_2439541424639
// MI455X (gfx1250) — compile-verified, matrix operand order checked by reading, not run on hardware
//
#include <hip/hip_runtime.h>
#include <hip/hip_bf16.h>

// BitLinear MLP on MI455X (gfx1250):
//   out = (int8 x_q  @ ternary w1) * sw1 / sx  -> gelu -> requant ->
//         (int8 h_q  @ ternary w2) * sw2 / sh
// GEMMs use V_WMMA_I32_16X16X64_IU8 (signed x signed) = exact reference math.
// Global->LDS staging uses GLOBAL_LOAD_ASYNC_TO_LDS_B128 (ASYNCcnt), the
// CDNA5 direct-to-LDS path: no VGPR transit, no register double-buffer.

typedef __attribute__((ext_vector_type(8))) int v8i;

#define TILE_M 128
#define TILE_N 128
#define TILE_K 128   // two WMMA K=64 sub-steps per LDS stage

__device__ __forceinline__ float gelu_f(float x) {
    const float c = 0.7978845608028654f;   // sqrt(2/pi)
    float x3 = x * x * x;
    return 0.5f * x * (1.0f + tanhf(c * (x + 0.044715f * x3)));
}

__device__ __forceinline__ v8i pack4(unsigned long long a, unsigned long long b,
                                     unsigned long long c, unsigned long long d) {
    union { v8i v; unsigned long long q[4]; } u;
    u.q[0] = a; u.q[1] = b; u.q[2] = c; u.q[3] = d;
    return u.v;
}

// 16-byte global -> LDS asynchronous copy (ASYNCcnt-tracked).
// VDST = LDS byte address (low 32 bits of the generic shared pointer,
// per the ISA aperture rule); VADDR = 64-bit global addr.
__device__ __forceinline__ void async_copy16(const signed char* g, signed char* l) {
    const unsigned lds_off = (unsigned)(size_t)l;
    asm volatile("global_load_async_to_lds_b128 %0, %1, off"
                 :: "v"(lds_off), "v"(g)
                 : "memory");
}

__device__ __forceinline__ void wait_async0() {
    asm volatile("s_wait_asynccnt 0x0" ::: "memory");
}

// ---------------------------------------------------------------------------
// Per-row ternary weight quantization: scale = max(mean|w|,1e-5),
// t = clamp(round(w/scale),-1,1) as int8; store scale per row.
// ---------------------------------------------------------------------------
__global__ __launch_bounds__(256)
void quant_weight_kernel(const float* __restrict__ w, signed char* __restrict__ wq,
                         float* __restrict__ scales, int K) {
    const int row  = blockIdx.x;
    const int tid  = threadIdx.x;
    const int lane = tid & 31;
    const int wid  = tid >> 5;
    const float* wr = w + (size_t)row * K;

    float s = 0.0f;
    for (int i = tid; i < K; i += 256) s += fabsf(wr[i]);
    #pragma unroll
    for (int off = 16; off > 0; off >>= 1) s += __shfl_xor(s, off, 32);

    __shared__ float red[8];
    if (lane == 0) red[wid] = s;
    __syncthreads();
    if (tid == 0) {
        float t = 0.0f;
        #pragma unroll
        for (int i = 0; i < 8; ++i) t += red[i];
        float scale = fmaxf(t / (float)K, 1e-5f);
        red[0] = scale;
        scales[row] = scale;
    }
    __syncthreads();
    const float inv = 1.0f / red[0];

    for (int i = tid; i < K; i += 256) {
        float q = rintf(wr[i] * inv);
        q = fminf(fmaxf(q, -1.0f), 1.0f);
        wq[(size_t)row * K + i] = (signed char)(int)q;
    }
}

// ---------------------------------------------------------------------------
// Per-row int8 activation quantization: sx = 127/max(absmax,1e-5),
// q = clamp(round(x*sx),-128,127); store inv_scale = max(absmax,1e-5)/127.
// ---------------------------------------------------------------------------
__global__ __launch_bounds__(256)
void quant_act_kernel(const float* __restrict__ x, signed char* __restrict__ xq,
                      float* __restrict__ inv_scale, int K) {
    const int row  = blockIdx.x;
    const int tid  = threadIdx.x;
    const int lane = tid & 31;
    const int wid  = tid >> 5;
    const float* xr = x + (size_t)row * K;

    float m = 0.0f;
    for (int i = tid; i < K; i += 256) m = fmaxf(m, fabsf(xr[i]));
    #pragma unroll
    for (int off = 16; off > 0; off >>= 1) m = fmaxf(m, __shfl_xor(m, off, 32));

    __shared__ float red[8];
    if (lane == 0) red[wid] = m;
    __syncthreads();
    if (tid == 0) {
        float t = 0.0f;
        #pragma unroll
        for (int i = 0; i < 8; ++i) t = fmaxf(t, red[i]);
        t = fmaxf(t, 1e-5f);
        red[0] = t;
        inv_scale[row] = t / 127.0f;
    }
    __syncthreads();
    const float s = 127.0f / red[0];

    for (int i = tid; i < K; i += 256) {
        float q = rintf(xr[i] * s);
        q = fminf(fmaxf(q, -128.0f), 127.0f);
        xq[(size_t)row * K + i] = (signed char)(int)q;
    }
}

// ---------------------------------------------------------------------------
// int8 x int8(ternary) GEMM, i32 accumulate via V_WMMA_I32_16X16X64_IU8.
//   A [M,K] row-major int8, B [N,K] row-major int8: C[m,n] = sum_k A[m,k]B[n,k]
// Epilogue: C_f32 = acc * scaleB[n] * invScaleA[m], optional GELU.
// 256 threads (8 wave32) per 128x128 tile; wave (wm 0..3, wn 0..1) owns
// 32(M) x 64(N) = 2x4 WMMA tiles. K staged 128 bytes deep through
// double-buffered LDS filled by async global->LDS copies.
// ---------------------------------------------------------------------------
__global__ __launch_bounds__(256)
void gemm_iu8_kernel(const signed char* __restrict__ Ag,
                     const signed char* __restrict__ Bg,
                     const float* __restrict__ scaleB,
                     const float* __restrict__ invScaleA,
                     float* __restrict__ C,
                     int M, int N, int K, int apply_gelu) {
    __shared__ __align__(16) signed char ldsA[2][TILE_M * TILE_K]; // 2 x 16 KB
    __shared__ __align__(16) signed char ldsB[2][TILE_N * TILE_K]; // 2 x 16 KB

    const int tid  = threadIdx.x;
    const int lane = tid & 31;
    const int wid  = tid >> 5;
    const int wm   = wid >> 1;      // 0..3 -> 32-row band
    const int wn   = wid & 1;       // 0..1 -> 64-col band
    const int m0   = blockIdx.y * TILE_M;
    const int n0   = blockIdx.x * TILE_N;

    const v8i vzero = {0, 0, 0, 0, 0, 0, 0, 0};
    v8i acc[2][4];
    #pragma unroll
    for (int mt = 0; mt < 2; ++mt)
        #pragma unroll
        for (int nt = 0; nt < 4; ++nt) acc[mt][nt] = vzero;

    // WMMA 8-bit operand byte mapping (ISA 7.12.2): lane = row (mod 16),
    // 8-byte K-groups at byte offsets {0,16,32,48} (+8 for lanes 16..31).
    const int fr   = lane & 15;
    const int kgrp = (lane >> 4) * 8;

    // Each stage moves A:128x128B + B:128x128B = 32 KB; 256 threads x 8
    // async b128 copies. Per-thread chunk ids: row = id>>3, seg = (id&7)*16.
    const int nstages = K / TILE_K;

    auto issue_stage = [&](int s, int buf) {
        const size_t kbase = (size_t)s * TILE_K;
        #pragma unroll
        for (int i = 0; i < 4; ++i) {
            const int id  = tid + i * 256;       // 0..1023
            const int row = id >> 3;             // 0..127
            const int seg = (id & 7) << 4;       // 0..112
            async_copy16(Ag + (size_t)(m0 + row) * K + kbase + seg,
                         &ldsA[buf][row * TILE_K + seg]);
        }
        #pragma unroll
        for (int i = 0; i < 4; ++i) {
            const int id  = tid + i * 256;
            const int row = id >> 3;
            const int seg = (id & 7) << 4;
            async_copy16(Bg + (size_t)(n0 + row) * K + kbase + seg,
                         &ldsB[buf][row * TILE_K + seg]);
        }
    };

    issue_stage(0, 0);

    for (int s = 0; s < nstages; ++s) {
        const int buf = s & 1;
        wait_async0();        // this wave's stage-s copies landed in LDS
        __syncthreads();      // ... and every other wave's too

        if (s + 1 < nstages) issue_stage(s + 1, buf ^ 1);

        #pragma unroll
        for (int ks = 0; ks < TILE_K; ks += 64) {
            v8i af[2];
            #pragma unroll
            for (int mt = 0; mt < 2; ++mt) {
                const signed char* p =
                    &ldsA[buf][(wm * 32 + mt * 16 + fr) * TILE_K + ks + kgrp];
                af[mt] = pack4(*(const unsigned long long*)(p),
                               *(const unsigned long long*)(p + 16),
                               *(const unsigned long long*)(p + 32),
                               *(const unsigned long long*)(p + 48));
            }
            v8i bf[4];
            #pragma unroll
            for (int nt = 0; nt < 4; ++nt) {
                const signed char* p =
                    &ldsB[buf][(wn * 64 + nt * 16 + fr) * TILE_K + ks + kgrp];
                bf[nt] = pack4(*(const unsigned long long*)(p),
                               *(const unsigned long long*)(p + 16),
                               *(const unsigned long long*)(p + 32),
                               *(const unsigned long long*)(p + 48));
            }
            #pragma unroll
            for (int mt = 0; mt < 2; ++mt)
                #pragma unroll
                for (int nt = 0; nt < 4; ++nt)
                    acc[mt][nt] = __builtin_amdgcn_wmma_i32_16x16x64_iu8(
                        /*sgn_a=*/true, af[mt], /*sgn_b=*/true, bf[nt],
                        acc[mt][nt], /*reuse_a=*/false, /*reuse_b=*/false);
        }

        __syncthreads();      // all reads of buf done before it is refilled
    }

    // Epilogue: dequant (+ GELU) and store.
    // C/D i32 16x16 layout: VGPR r holds M = r + 8*(lane>=16), N = lane&15.
    #pragma unroll
    for (int mt = 0; mt < 2; ++mt) {
        const int mbase = m0 + wm * 32 + mt * 16 + (lane >> 4) * 8;
        #pragma unroll
        for (int nt = 0; nt < 4; ++nt) {
            const int col = n0 + wn * 64 + nt * 16 + fr;
            const float sB = scaleB[col];
            #pragma unroll
            for (int r = 0; r < 8; ++r) {
                const int row = mbase + r;
                float v = (float)acc[mt][nt][r] * sB * invScaleA[row];
                if (apply_gelu) v = gelu_f(v);
                C[(size_t)row * N + col] = v;
            }
        }
    }
}

// ---------------------------------------------------------------------------
// Host-side launcher
// ---------------------------------------------------------------------------
extern "C" void kernel_launch(void* const* d_in, const int* in_sizes, int n_in,
                              void* d_out, int out_size, void* d_ws, size_t ws_size,
                              hipStream_t stream) {
    const float* x  = (const float*)d_in[0];   // [B,S,D] = [4,2048,2048]
    const float* w1 = (const float*)d_in[1];   // [H,D]   = [8192,2048]
    const float* w2 = (const float*)d_in[2];   // [D,H]   = [2048,8192]
    float* out = (float*)d_out;                // [B,S,D]

    const int D = 2048;
    const int H = 8192;
    const int M = in_sizes[0] / D;             // B*S = 8192

    char* base = (char*)d_ws;
    size_t off = 0;
    auto alloc = [&](size_t bytes) -> char* {
        char* p = base + off;
        off += (bytes + 255) & ~(size_t)255;
        return p;
    };
    signed char* w1q = (signed char*)alloc((size_t)H * D);      // 16 MB
    signed char* w2q = (signed char*)alloc((size_t)D * H);      // 16 MB
    signed char* xq  = (signed char*)alloc((size_t)M * D);      // 16 MB
    signed char* hq  = (signed char*)alloc((size_t)M * H);      // 64 MB
    float* sw1 = (float*)alloc((size_t)H * sizeof(float));
    float* sw2 = (float*)alloc((size_t)D * sizeof(float));
    float* isx = (float*)alloc((size_t)M * sizeof(float));
    float* ish = (float*)alloc((size_t)M * sizeof(float));
    float* hf  = (float*)alloc((size_t)M * H * sizeof(float));  // 256 MB

    quant_weight_kernel<<<H, 256, 0, stream>>>(w1, w1q, sw1, D);
    quant_weight_kernel<<<D, 256, 0, stream>>>(w2, w2q, sw2, H);
    quant_act_kernel<<<M, 256, 0, stream>>>(x, xq, isx, D);

    // Layer 1: [M,D] x [H,D]^T -> gelu -> hf [M,H]
    gemm_iu8_kernel<<<dim3(H / TILE_N, M / TILE_M), 256, 0, stream>>>(
        xq, w1q, sw1, isx, hf, M, H, D, /*gelu=*/1);

    quant_act_kernel<<<M, 256, 0, stream>>>(hf, hq, ish, H);

    // Layer 2: [M,H] x [D,H]^T -> out [M,D]
    gemm_iu8_kernel<<<dim3(D / TILE_N, M / TILE_M), 256, 0, stream>>>(
        hq, w2q, sw2, ish, out, M, D, H, /*gelu=*/0);
}
